// MOE_87007447482599
// MI455X (gfx1250) — compile-verified, matrix operand order checked by reading, not run on hardware
//
#include <hip/hip_runtime.h>
#include <math.h>

// ---------------- problem constants ----------------
constexpr int Bc = 4, Sc = 2048, Dc = 1024;
constexpr int Ec = 8, Kc = 2, Ic = 3072;
constexpr int Tc = Bc * Sc;          // 8192 tokens
constexpr int TRc = Tc * Kc;         // 16384 expanded rows
constexpr int TRPc = TRc + Ec * 16;  // 16512 padded capacity (each expert 16-aligned)
constexpr int APAD = 8;              // LDS row pad (16B) -> conflict-free ds_load_b128

// ---------------- vector types ----------------
typedef __attribute__((ext_vector_type(16))) __bf16 v16bf;
typedef __attribute__((ext_vector_type(8)))  float  v8f;
typedef int v4i __attribute__((vector_size(16)));
typedef __attribute__((address_space(1))) v4i* gv4i_p;
typedef __attribute__((address_space(3))) v4i* lv4i_p;

// float -> bf16 (round-to-nearest-even), stored as raw u16
__device__ __forceinline__ unsigned short f2bf(float f) {
    unsigned int u = __builtin_bit_cast(unsigned int, f);
    unsigned int r = u + 0x7FFFu + ((u >> 16) & 1u);
    return (unsigned short)(r >> 16);
}

// ---------------- async stage: 16 x KD bf16 row tile -> LDS ----------------
template <int KD>
__device__ __forceinline__ void stage_a(const unsigned short* __restrict__ g,
                                        unsigned short* l) {
    const int tid = threadIdx.x;
    const int chunks_per_row = KD / 8;        // 16-byte chunks per row
    const int total = 16 * chunks_per_row;
    for (int c = tid; c < total; c += (int)blockDim.x) {
        int row = c / chunks_per_row;
        int off = (c % chunks_per_row) * 8;
        const unsigned short* gs = g + (size_t)row * KD + off;
        unsigned short* ls = l + row * (KD + APAD) + off;
#if __has_builtin(__builtin_amdgcn_global_load_async_to_lds_b128)
        __builtin_amdgcn_global_load_async_to_lds_b128(
            (gv4i_p)(unsigned short*)gs, (lv4i_p)ls, 0, 0);
#else
        *(uint4*)ls = *(const uint4*)gs;
#endif
    }
#if __has_builtin(__builtin_amdgcn_global_load_async_to_lds_b128)
#if __has_builtin(__builtin_amdgcn_s_wait_asynccnt)
    __builtin_amdgcn_s_wait_asynccnt(0);
#else
    asm volatile("s_wait_asynccnt 0x0" ::: "memory");
#endif
#endif
    __syncthreads();
}

// ---------------- init kernels ----------------
__global__ void zero_f32_k(float* p, size_t n) {
    for (size_t i = (size_t)blockIdx.x * blockDim.x + threadIdx.x; i < n;
         i += (size_t)gridDim.x * blockDim.x) p[i] = 0.0f;
}
__global__ void zero_u16_k(unsigned short* p, size_t n) {
    for (size_t i = (size_t)blockIdx.x * blockDim.x + threadIdx.x; i < n;
         i += (size_t)gridDim.x * blockDim.x) p[i] = 0;
}
__global__ void small_init_k(int* row_map, float* gatew, int* counts,
                             float* probsum, float* z2sum) {
    int i = blockIdx.x * blockDim.x + threadIdx.x;
    if (i < TRPc) { row_map[i] = -1; gatew[i] = 0.0f; }
    if (i < Ec)   { counts[i] = 0;   probsum[i] = 0.0f; }
    if (i == 0)   { z2sum[0] = 0.0f; }
}

// fp32 [E][R][C] -> bf16 [E][C][R]  (transpose within each expert)
__global__ void cvt_transpose_k(const float* __restrict__ src,
                                unsigned short* __restrict__ dst,
                                int R, int C) {
    size_t per = (size_t)R * C;
    size_t total = (size_t)Ec * per;
    for (size_t i = (size_t)blockIdx.x * blockDim.x + threadIdx.x; i < total;
         i += (size_t)gridDim.x * blockDim.x) {
        size_t e = i / per;
        size_t rem = i - e * per;
        int r = (int)(rem / C);
        int c = (int)(rem % C);
        dst[e * per + (size_t)c * R + r] = f2bf(src[i]);
    }
}

// ---------------- router: one wave per token ----------------
__global__ void router_k(const float* __restrict__ x, const float* __restrict__ rw,
                         int* __restrict__ topk_i, float* __restrict__ topk_w,
                         int* __restrict__ counts, float* __restrict__ probsum,
                         float* __restrict__ z2sum) {
    int wave = threadIdx.x >> 5, lane = threadIdx.x & 31;
    int t = blockIdx.x * (blockDim.x >> 5) + wave;
    if (t >= Tc) return;
    const float* xr = x + (size_t)t * Dc;
    float acc[Ec];
#pragma unroll
    for (int e = 0; e < Ec; ++e) acc[e] = 0.0f;
    for (int d = lane; d < Dc; d += 32) {
        float xv = xr[d];
        const float* rr = rw + (size_t)d * Ec;
#pragma unroll
        for (int e = 0; e < Ec; ++e) acc[e] += xv * rr[e];
    }
#pragma unroll
    for (int e = 0; e < Ec; ++e) {
        float v = acc[e];
#pragma unroll
        for (int o = 16; o >= 1; o >>= 1) v += __shfl_xor(v, o, 32);
        acc[e] = v;
    }
    if (lane == 0) {
        // top-2, first-occurrence tie-break (matches lax.top_k)
        int i1 = 0;
#pragma unroll
        for (int e = 1; e < Ec; ++e) if (acc[e] > acc[i1]) i1 = e;
        int i2 = (i1 == 0) ? 1 : 0;
#pragma unroll
        for (int e = 0; e < Ec; ++e) if (e != i1 && acc[e] > acc[i2]) i2 = e;
        float m = acc[i1];
        float p[Ec], s = 0.0f;
#pragma unroll
        for (int e = 0; e < Ec; ++e) { p[e] = __expf(acc[e] - m); s += p[e]; }
        float inv = 1.0f / s;
#pragma unroll
        for (int e = 0; e < Ec; ++e) {
            p[e] *= inv;
            __hip_atomic_fetch_add(&probsum[e], p[e], __ATOMIC_RELAXED,
                                   __HIP_MEMORY_SCOPE_AGENT);
        }
        float z = m + __logf(s);
        __hip_atomic_fetch_add(z2sum, z * z, __ATOMIC_RELAXED,
                               __HIP_MEMORY_SCOPE_AGENT);
        atomicAdd(&counts[i1], 1);
        atomicAdd(&counts[i2], 1);
        float wn = 1.0f / (p[i1] + p[i2]);
        topk_i[2 * t]     = i1;  topk_w[2 * t]     = p[i1] * wn;
        topk_i[2 * t + 1] = i2;  topk_w[2 * t + 1] = p[i2] * wn;
    }
}

// ---------------- padded exclusive prefix over expert counts ----------------
__global__ void offsets_k(const int* __restrict__ counts, int* __restrict__ offs,
                          int* __restrict__ cursor) {
    if (threadIdx.x == 0) {
        int o = 0;
        for (int e = 0; e < Ec; ++e) {
            offs[e] = o;
            o += (counts[e] + 15) & ~15;   // 16-align each expert group
        }
        offs[Ec] = o;
    }
    if (threadIdx.x < Ec) cursor[threadIdx.x] = 0;
}

// ---------------- scatter: one wave per (token, k) ----------------
__global__ void scatter_k(const float* __restrict__ x, const int* __restrict__ topk_i,
                          const float* __restrict__ topk_w, const int* __restrict__ offs,
                          int* __restrict__ cursor, int* __restrict__ row_map,
                          float* __restrict__ gatew, unsigned short* __restrict__ xp) {
    int wave = threadIdx.x >> 5, lane = threadIdx.x & 31;
    int idx = blockIdx.x * (blockDim.x >> 5) + wave;
    if (idx >= TRc) return;
    int t = idx >> 1;
    int pos = 0;
    if (lane == 0) {
        int e = topk_i[idx];
        pos = offs[e] + atomicAdd(&cursor[e], 1);
        row_map[pos] = t;
        gatew[pos] = topk_w[idx];
    }
    pos = __shfl(pos, 0, 32);
    const float* xr = x + (size_t)t * Dc;
    unsigned short* dst = xp + (size_t)pos * Dc;
    for (int d = lane; d < Dc; d += 32) dst[d] = f2bf(xr[d]);
}

// ---------------- GEMM1: gated = silu(xp·w1) * (xp·w2), bf16 WMMA ----------------
// block: 4 waves share one 16-row A tile (LDS-staged); each wave owns 64 I-cols.
__global__ void gemm1_k(const unsigned short* __restrict__ xp,
                        const unsigned short* __restrict__ w1t,   // [E][I][D] bf16
                        const unsigned short* __restrict__ w2t,   // [E][I][D] bf16
                        const int* __restrict__ offs,
                        unsigned short* __restrict__ gated) {     // [TRP][I] bf16
    __shared__ unsigned short sA[16 * (Dc + APAD)];
    const int wave = threadIdx.x >> 5, lane = threadIdx.x & 31;
    const int row0 = blockIdx.x * 16;
    const int total = offs[Ec];
    if (row0 >= total) return;
    int e = 0;
#pragma unroll
    for (int q = 1; q < Ec; ++q) if (row0 >= offs[q]) e = q;

    stage_a<Dc>(xp + (size_t)row0 * Dc, sA);

    const int col0 = (blockIdx.y * 4 + wave) * 64;
    const int half = lane >> 4, l16 = lane & 15;

    const unsigned short* aL = sA + l16 * (Dc + APAD) + half * 16;
    const size_t wbase = (size_t)e * Ic * Dc + (size_t)(col0 + l16) * Dc + half * 16;
    const unsigned short* b1 = w1t + wbase;
    const unsigned short* b2 = w2t + wbase;

    v8f acc1[4] = {}, acc2[4] = {};
    for (int kk = 0; kk < Dc; kk += 32) {
        v16bf a = *reinterpret_cast<const v16bf*>(aL + kk);
        v16bf vb[8];
#pragma unroll
        for (int j = 0; j < 4; ++j) {
            vb[j]     = *reinterpret_cast<const v16bf*>(b1 + (size_t)j * 16 * Dc + kk);
            vb[4 + j] = *reinterpret_cast<const v16bf*>(b2 + (size_t)j * 16 * Dc + kk);
        }
#pragma unroll
        for (int j = 0; j < 4; ++j) {
            acc1[j] = __builtin_amdgcn_wmma_f32_16x16x32_bf16(
                false, a, false, vb[j], (short)0, acc1[j], false, false);
            acc2[j] = __builtin_amdgcn_wmma_f32_16x16x32_bf16(
                false, a, false, vb[4 + j], (short)0, acc2[j], false, false);
        }
    }
#pragma unroll
    for (int j = 0; j < 4; ++j) {
        int col = col0 + j * 16 + l16;
#pragma unroll
        for (int r = 0; r < 8; ++r) {
            int row = row0 + r + 8 * half;
            float h1 = acc1[j][r], h2 = acc2[j][r];
            float g = (h1 / (1.0f + __expf(-h1))) * h2;   // silu(h1)*h2
            gated[(size_t)row * Ic + col] = f2bf(g);
        }
    }
}

// ---------------- GEMM2: out += gate * (gated · w3), atomic combine ----------------
__global__ void gemm2_k(const unsigned short* __restrict__ gated,  // [TRP][I] bf16
                        const unsigned short* __restrict__ w3t,    // [E][D][I] bf16
                        const int* __restrict__ offs,
                        const int* __restrict__ row_map,
                        const float* __restrict__ gatew,
                        float* __restrict__ out) {                 // [T][D] f32
    __shared__ unsigned short sA[16 * (Ic + APAD)];
    const int wave = threadIdx.x >> 5, lane = threadIdx.x & 31;
    const int row0 = blockIdx.x * 16;
    const int total = offs[Ec];
    if (row0 >= total) return;
    int e = 0;
#pragma unroll
    for (int q = 1; q < Ec; ++q) if (row0 >= offs[q]) e = q;

    stage_a<Ic>(gated + (size_t)row0 * Ic, sA);

    const int col0 = (blockIdx.y * 4 + wave) * 64;
    const int half = lane >> 4, l16 = lane & 15;

    const unsigned short* aL = sA + l16 * (Ic + APAD) + half * 16;
    const unsigned short* b3 =
        w3t + (size_t)e * Dc * Ic + (size_t)(col0 + l16) * Ic + half * 16;

    v8f acc[4] = {};
    for (int kk = 0; kk < Ic; kk += 32) {
        v16bf a = *reinterpret_cast<const v16bf*>(aL + kk);
        v16bf vb[4];
#pragma unroll
        for (int j = 0; j < 4; ++j)
            vb[j] = *reinterpret_cast<const v16bf*>(b3 + (size_t)j * 16 * Ic + kk);
#pragma unroll
        for (int j = 0; j < 4; ++j)
            acc[j] = __builtin_amdgcn_wmma_f32_16x16x32_bf16(
                false, a, false, vb[j], (short)0, acc[j], false, false);
    }
#pragma unroll
    for (int r = 0; r < 8; ++r) {
        int row = row0 + r + 8 * half;
        int t = row_map[row];
        if (t < 0) continue;
        float g = gatew[row];
#pragma unroll
        for (int j = 0; j < 4; ++j) {
            int col = col0 + j * 16 + l16;
            __hip_atomic_fetch_add(&out[(size_t)t * Dc + col], acc[j][r] * g,
                                   __ATOMIC_RELAXED, __HIP_MEMORY_SCOPE_AGENT);
        }
    }
}

// ---------------- aux losses ----------------
__global__ void finalize_k(const int* __restrict__ counts,
                           const float* __restrict__ probsum,
                           const float* __restrict__ z2sum,
                           float* __restrict__ tail) {
    if (blockIdx.x == 0 && threadIdx.x == 0) {
        float lbl = 0.0f;
        for (int e = 0; e < Ec; ++e) {
            float load = (float)counts[e] / (float)TRc;
            tail[e] = load;
            float f = load * ((float)Ec / (float)Kc);
            float pm = probsum[e] / (float)Tc;
            lbl += f * pm;
        }
        tail[Ec] = 0.01f * lbl;
        tail[Ec + 1] = 0.001f * (z2sum[0] / (float)Tc);
    }
}

// ---------------- host launch ----------------
extern "C" void kernel_launch(void* const* d_in, const int* in_sizes, int n_in,
                              void* d_out, int out_size, void* d_ws, size_t ws_size,
                              hipStream_t stream) {
    const float* x  = (const float*)d_in[0];
    const float* rw = (const float*)d_in[1];
    const float* w1 = (const float*)d_in[2];
    const float* w2 = (const float*)d_in[3];
    const float* w3 = (const float*)d_in[4];
    float* out = (float*)d_out;                 // [T*D] then [E] loads, lbl, rzl
    float* tail = out + (size_t)Tc * Dc;

    // workspace carve-up (~287 MB total)
    constexpr size_t SZ_W  = (size_t)Ec * Dc * Ic * 2;       // bf16 weight copy
    constexpr size_t SZ_XP = (size_t)TRPc * Dc * 2;
    constexpr size_t SZ_G  = (size_t)TRPc * Ic * 2;
    char* base = (char*)d_ws;
    unsigned short* w1t   = (unsigned short*)(base);
    unsigned short* w2t   = (unsigned short*)(base + SZ_W);
    unsigned short* w3t   = (unsigned short*)(base + 2 * SZ_W);
    unsigned short* xp    = (unsigned short*)(base + 3 * SZ_W);
    unsigned short* gated = (unsigned short*)(base + 3 * SZ_W + SZ_XP);
    char* p = base + 3 * SZ_W + SZ_XP + SZ_G;
    int*   topk_i  = (int*)p;    p += (size_t)TRc * 4;
    float* topk_w  = (float*)p;  p += (size_t)TRc * 4;
    int*   row_map = (int*)p;    p += (size_t)TRPc * 4;
    float* gatew   = (float*)p;  p += (size_t)TRPc * 4;
    int*   counts  = (int*)p;    p += 64;
    int*   offs    = (int*)p;    p += 64;
    int*   cursor  = (int*)p;    p += 64;
    float* probsum = (float*)p;  p += 64;
    float* z2sum   = (float*)p;

    // 1) init output + pad rows + scalars
    zero_f32_k<<<dim3(2048), dim3(256), 0, stream>>>(out, (size_t)Tc * Dc);
    zero_u16_k<<<dim3(2048), dim3(256), 0, stream>>>(xp, (size_t)TRPc * Dc);
    small_init_k<<<dim3((TRPc + 255) / 256), dim3(256), 0, stream>>>(
        row_map, gatew, counts, probsum, z2sum);

    // 2) fp32 -> bf16 transposed weights
    cvt_transpose_k<<<dim3(4096), dim3(256), 0, stream>>>(w1, w1t, Dc, Ic);
    cvt_transpose_k<<<dim3(4096), dim3(256), 0, stream>>>(w2, w2t, Dc, Ic);
    cvt_transpose_k<<<dim3(4096), dim3(256), 0, stream>>>(w3, w3t, Ic, Dc);

    // 3) router (one wave per token)
    router_k<<<dim3(Tc / 8), dim3(256), 0, stream>>>(x, rw, topk_i, topk_w,
                                                     counts, probsum, z2sum);
    // 4) padded prefix sum
    offsets_k<<<dim3(1), dim3(64), 0, stream>>>(counts, offs, cursor);

    // 5) scatter rows to bf16 (one wave per expanded row)
    scatter_k<<<dim3(TRc / 8), dim3(256), 0, stream>>>(x, topk_i, topk_w, offs,
                                                       cursor, row_map, gatew, xp);

    // 6) grouped GEMM1 + SwiGLU  (grid: row tiles x col super-tiles)
    gemm1_k<<<dim3(TRPc / 16, Ic / 256), dim3(128), 0, stream>>>(
        xp, w1t, w2t, offs, gated);

    // 7) grouped GEMM2 + gated atomic combine
    gemm2_k<<<dim3(TRPc / 16, Dc / 256), dim3(128), 0, stream>>>(
        gated, w3t, offs, row_map, gatew, out);

    // 8) aux losses
    finalize_k<<<dim3(1), dim3(1), 0, stream>>>(counts, probsum, z2sum, tail);
}
